// NeuralMemory_65506841198866
// MI455X (gfx1250) — compile-verified, matrix operand order checked by reading, not run on hardware
//
#include <hip/hip_runtime.h>
#include <hip/hip_bf16.h>

// Output = retrieved = (x @ Wk^T) @ memory.  (q, v, grad, opt updates are dead code.)
// bf16 WMMA (f32 accumulate), fused double-GEMM; k lives only in LDS; output
// tile is transposed through LDS so global stores are coalesced b128.

typedef __attribute__((ext_vector_type(16))) __bf16         v16bf;
typedef __attribute__((ext_vector_type(8)))  float          v8f;
typedef __attribute__((ext_vector_type(8)))  unsigned short v8u16;

union ABu { v16bf bf; struct { v8u16 lo, hi; } h; };

static __device__ __forceinline__ unsigned short f2bf(float f) {
    unsigned int u = __float_as_uint(f);
    u = (u + 0x7FFFu + ((u >> 16) & 1u)) >> 16;   // round-to-nearest-even
    return (unsigned short)u;
}

// ---- pre-pass: Wk f32[128][512] -> bf16 row-major; memory f32[128][512] -> bf16 transposed [512][128]
__global__ __launch_bounds__(256) void nm_cvt_weights(
    const float* __restrict__ Wk, const float* __restrict__ mem,
    unsigned short* __restrict__ wk_b, unsigned short* __restrict__ memT)
{
    int i = blockIdx.x * 256 + threadIdx.x;      // 0 .. 65535
    if (i < 128 * 512) {
        wk_b[i] = f2bf(Wk[i]);
        int m = i >> 9;          // row in memory
        int d = i & 511;         // col in memory
        memT[d * 128 + m] = f2bf(mem[i]);
    }
}

#define XS 520   // x tile row stride: 512 + 8 bf16 pad (1040 B, 16B-aligned, conflict-free)
#define KS 136   // k tile row stride: 128 + 8 bf16 pad (272 B)
#define OS 516   // out tile row stride: 512 + 4 f32 pad (2064 B, 16B-aligned, conflict-free)

__global__ __launch_bounds__(256) void nm_fused_gemm(
    const float* __restrict__ x,                 // [32768, 512] f32
    const unsigned short* __restrict__ wk_b,     // [128, 512]  bf16
    const unsigned short* __restrict__ memT,     // [512, 128]  bf16 (memory^T)
    float* __restrict__ out)                     // [32768, 512] f32
{
    // out_s (33024 B) aliases xs (16640 B): xs is dead after the post-GEMM1 barrier.
    __shared__ __align__(16) unsigned char smem[16 * OS * 4 + 16 * KS * 2];
    unsigned short* xs    = (unsigned short*)smem;               // [16][XS] bf16
    float*          out_s = (float*)smem;                        // [16][OS] f32
    unsigned short* ks    = (unsigned short*)(smem + 16 * OS * 4); // [16][KS] bf16

    const int tile = blockIdx.x;                 // 2048 tiles of 16 rows
    const int t    = threadIdx.x;
    const int lane = t & 31;
    const int w    = t >> 5;                     // wave 0..7
    const int half = lane >> 4;                  // 0 | 1
    const int l16  = lane & 15;

    // ---- stage x tile (16x512 f32) into LDS as bf16, 32 elements/thread ----
    {
        const int row = t >> 4;                  // 0..15
        const int cb  = (t & 15) * 32;           // 0,32,..,480
        const float4* src = (const float4*)(x + ((size_t)(tile * 16 + row) * 512 + cb));
        unsigned short* dst = xs + row * XS + cb;
        #pragma unroll
        for (int j = 0; j < 4; ++j) {
            float4 f0 = src[2 * j];
            float4 f1 = src[2 * j + 1];
            v8u16 p;
            p[0] = f2bf(f0.x); p[1] = f2bf(f0.y); p[2] = f2bf(f0.z); p[3] = f2bf(f0.w);
            p[4] = f2bf(f1.x); p[5] = f2bf(f1.y); p[6] = f2bf(f1.z); p[7] = f2bf(f1.w);
            *(v8u16*)(dst + j * 8) = p;
        }
    }
    __syncthreads();

    // ---- GEMM1: k[16 x 16] (this wave's N-chunk) = x_tile[16x512] @ Wk^T ----
    v8f c1 = {0.f, 0.f, 0.f, 0.f, 0.f, 0.f, 0.f, 0.f};
    #pragma unroll
    for (int kc = 0; kc < 16; ++kc) {
        const int co = kc * 32 + 8 * half;       // per-lane K-run start (ISA 16-bit A/B layout)
        ABu a, b;
        a.h.lo = *(const v8u16*)(xs + l16 * XS + co);
        a.h.hi = *(const v8u16*)(xs + l16 * XS + co + 16);
        const unsigned short* wrow = wk_b + (size_t)(w * 16 + l16) * 512 + co;
        b.h.lo = *(const v8u16*)(wrow);
        b.h.hi = *(const v8u16*)(wrow + 16);
        c1 = __builtin_amdgcn_wmma_f32_16x16x32_bf16(false, a.bf, false, b.bf,
                                                     (short)0, c1, false, false);
    }
    // C-frag (VGPR r <-> row r + 8*half, col = lane%16) -> k tile in LDS as bf16
    #pragma unroll
    for (int r = 0; r < 8; ++r)
        ks[(r + 8 * half) * KS + w * 16 + l16] = f2bf(c1[r]);
    __syncthreads();   // also retires all xs reads -> out_s may reuse that space

    // ---- GEMM2: out cols [64w, 64w+64) = k[16x128] @ memory[128x512] ----
    v8f acc0 = {0.f,0.f,0.f,0.f,0.f,0.f,0.f,0.f};
    v8f acc1 = acc0, acc2 = acc0, acc3 = acc0;
    #pragma unroll
    for (int kc = 0; kc < 4; ++kc) {
        const int co = kc * 32 + 8 * half;
        ABu a;
        a.h.lo = *(const v8u16*)(ks + l16 * KS + co);
        a.h.hi = *(const v8u16*)(ks + l16 * KS + co + 16);
        #pragma unroll
        for (int nc = 0; nc < 4; ++nc) {
            ABu b;
            const unsigned short* mrow = memT + (size_t)((w * 4 + nc) * 16 + l16) * 128 + co;
            b.h.lo = *(const v8u16*)(mrow);
            b.h.hi = *(const v8u16*)(mrow + 16);
            v8f cin = (nc == 0) ? acc0 : (nc == 1) ? acc1 : (nc == 2) ? acc2 : acc3;
            v8f cres = __builtin_amdgcn_wmma_f32_16x16x32_bf16(false, a.bf, false, b.bf,
                                                               (short)0, cin, false, false);
            if      (nc == 0) acc0 = cres;
            else if (nc == 1) acc1 = cres;
            else if (nc == 2) acc2 = cres;
            else              acc3 = cres;
        }
    }

    // ---- transpose result tile through LDS ----
    #pragma unroll
    for (int nc = 0; nc < 4; ++nc) {
        const v8f accv = (nc == 0) ? acc0 : (nc == 1) ? acc1 : (nc == 2) ? acc2 : acc3;
        const int col = (w * 4 + nc) * 16 + l16;
        #pragma unroll
        for (int r = 0; r < 8; ++r)
            out_s[(r + 8 * half) * OS + col] = accv[r];
    }
    __syncthreads();

    // ---- cooperative coalesced store: consecutive lanes -> consecutive 16B ----
    {
        float* obase = out + (size_t)tile * (16 * 512);
        #pragma unroll
        for (int j = 0; j < 8; ++j) {
            const int f   = (j * 256 + t) * 4;   // dword offset within 16x512 tile
            const int row = f >> 9;
            const int col = f & 511;
            float4 val = *(const float4*)(out_s + row * OS + col);
            *(float4*)(obase + f) = val;
        }
    }
}

extern "C" void kernel_launch(void* const* d_in, const int* in_sizes, int n_in,
                              void* d_out, int out_size, void* d_ws, size_t ws_size,
                              hipStream_t stream) {
    (void)in_sizes; (void)n_in; (void)out_size; (void)ws_size;
    const float* x   = (const float*)d_in[0];   // [8,4096,512]
    const float* Wk  = (const float*)d_in[2];   // [128,512]
    const float* mem = (const float*)d_in[4];   // [128,512]
    float* out = (float*)d_out;                 // [8,4096,512]

    unsigned short* wk_b = (unsigned short*)d_ws;        // 128*512*2 = 128 KB
    unsigned short* memT = wk_b + 128 * 512;             // another 128 KB

    nm_cvt_weights<<<256, 256, 0, stream>>>(Wk, mem, wk_b, memT);
    nm_fused_gemm<<<2048, 256, 0, stream>>>(x, wk_b, memT, out);
}
